// TernaryConv2d_19396072309134
// MI455X (gfx1250) — matrix operand order repaired and checked by reading, not compiled, not run on hardware
//
#include <hip/hip_runtime.h>

typedef _Float16 v8h  __attribute__((ext_vector_type(8)));
typedef _Float16 v16h __attribute__((ext_vector_type(16)));
typedef float    v8f  __attribute__((ext_vector_type(8)));

static constexpr int CIN   = 128;
static constexpr int COUT  = 256;
static constexpr int IH    = 64;
static constexpr int IW    = 64;
static constexpr int KDIM  = CIN * 9;        // 1152, k = (kh*3+kw)*128 + ci
static constexpr int WELEMS = COUT * KDIM;   // 294912

static constexpr int TX   = 34;              // staged x positions: x0-1 .. x0+32
static constexpr int CIP  = 136;             // ci padded 128 -> 136 halves (bank spread)
static constexpr int TILE = 3 * TX * CIP;    // 13872 halves = 27744 B

// ---------------------------------------------------------------------------
// Kernel 1: delta = 0.7 * mean(|w|), single block -> deterministic
// ---------------------------------------------------------------------------
__global__ void __launch_bounds__(256)
delta_kernel(const float* __restrict__ w, float* __restrict__ ws) {
    __shared__ float red[256];
    float s = 0.f;
    for (int i = threadIdx.x; i < WELEMS; i += 256) s += fabsf(w[i]);
    red[threadIdx.x] = s;
    __syncthreads();
    for (int off = 128; off > 0; off >>= 1) {
        if ((int)threadIdx.x < off) red[threadIdx.x] += red[threadIdx.x + off];
        __syncthreads();
    }
    if (threadIdx.x == 0) ws[0] = 0.7f * red[0] / (float)WELEMS;
}

// ---------------------------------------------------------------------------
// Kernel 2: ternarize + pack weight into WMMA-B-friendly layout wt[co][k]
//           k = (kh*3+kw)*128 + ci   (each 32-wide K step has fixed kh,kw)
// ---------------------------------------------------------------------------
__global__ void __launch_bounds__(256)
tern_kernel(const float* __restrict__ w, const float* __restrict__ ws,
            _Float16* __restrict__ wt) {
    int o = blockIdx.x * 256 + threadIdx.x;      // o = co*KDIM + k
    if (o >= WELEMS) return;
    float delta = ws[0];
    int co   = o / KDIM;
    int k    = o - co * KDIM;
    int khkw = k >> 7;           // 0..8
    int ci   = k & 127;
    float v  = w[(co * CIN + ci) * 9 + khkw];
    float t  = (v > delta) ? 1.f : ((v < -delta) ? -1.f : 0.f);
    wt[o] = (_Float16)t;
}

// ---------------------------------------------------------------------------
// Kernel 3: implicit-GEMM conv via v_wmma_f32_16x16x32_f16
//   block: 256 threads = 8 waves; tile = 32 pixels (n,y fixed; x0..x0+31)
//          x 256 output channels. Wave wv owns 32 pixels x 32 channels
//          (2x2 register blocking -> 4 accumulators, A/B fragment reuse).
//   LDS holds the RAW x tile once: xt[kh(3)][xx(34)][ci(136 padded)] f16;
//   the 3x3 window shift is folded into constant ds_load_b128 offsets of a
//   fully-unrolled 36-step K loop (all index math constant-folded).
// ---------------------------------------------------------------------------
__global__ void __launch_bounds__(256)
conv_wmma_kernel(const float* __restrict__ x, const _Float16* __restrict__ wt,
                 const float* __restrict__ bias, float* __restrict__ out) {
    __shared__ __align__(16) _Float16 Ah[TILE];  // 27744 B; reused for C transpose

    const int tid = threadIdx.x;
    unsigned bid  = blockIdx.x;
    const int x0  = (bid & 1) << 5;              // 2 x-halves of 32
    const int y   = (bid >> 1) & 63;
    const int n   = bid >> 7;                    // 0..31

    const int lane = tid & 31;
    const int wv   = tid >> 5;
    const int g    = lane >> 4;                  // half-wave select
    const int mrow = lane & 15;
    const int co0  = wv * 32;                    // 8 waves cover all 256 channels

    const float* xn = x + (size_t)n * (CIN * IH * IW);

    // ---- stage raw x tile: rows y-1..y+1, x0-1..x0+32, all 128 ci ----------
    for (int e = tid; e < 3 * TX * CIN; e += 256) {  // 13056 elems, 51 iters
        int xxp  = e % TX;                       // fast index -> coalesced x
        int rest = e / TX;
        int ci   = rest & 127;
        int kh   = rest >> 7;                    // 0..2
        int yy   = y + kh - 1;
        int xx   = x0 - 1 + xxp;
        float v  = 0.f;
        if ((unsigned)yy < 64u && (unsigned)xx < 64u)
            v = xn[(ci * IH + yy) * IW + xx];
        Ah[(kh * TX + xxp) * CIP + ci] = (_Float16)v;
    }
    __syncthreads();

    // per-lane fragment bases (A rows for the two m-tiles; B rows per co-tile)
    const _Float16* AL0 = Ah + mrow * CIP + 8 * g;         // pixel row = lane&15
    const _Float16* AL1 = AL0 + 16 * CIP;                  // pixel row + 16
    const _Float16* B0  = wt + (size_t)(co0 + mrow) * KDIM + 8 * g;
    const _Float16* B1  = B0 + (size_t)16 * KDIM;

    v8f acc00 = {}, acc01 = {}, acc10 = {}, acc11 = {};

#pragma unroll
    for (int s = 0; s < 36; ++s) {               // K = 1152 in 32-wide steps
        const int khkw = s >> 2;                 // constant-folded (full unroll)
        const int kh   = khkw / 3;
        const int kw   = khkw - kh * 3;
        const int ci0  = (s & 3) * 32;
        const int cs   = (kh * TX + kw) * CIP + ci0;  // step-constant LDS offset

        // ISA 16-bit A 16x32 layout: lanes 0-15 hold K 0-7 & 16-23,
        // lanes 16-31 hold K 8-15 & 24-31; K-contiguity == ci-contiguity here
        v8h a0lo = *(const v8h*)(AL0 + cs);
        v8h a0hi = *(const v8h*)(AL0 + cs + 16);
        v8h a1lo = *(const v8h*)(AL1 + cs);
        v8h a1hi = *(const v8h*)(AL1 + cs + 16);
        v16h a0 = __builtin_shufflevector(a0lo, a0hi,
                    0,1,2,3,4,5,6,7,8,9,10,11,12,13,14,15);
        v16h a1 = __builtin_shufflevector(a1lo, a1hi,
                    0,1,2,3,4,5,6,7,8,9,10,11,12,13,14,15);
        // B 32x16: lane n holds column n, K s*32+8g..+7 and s*32+16+8g..+7
        v16h b0 = __builtin_shufflevector(*(const v8h*)(B0 + s * 32), *(const v8h*)(B0 + s * 32 + 16),
                    0,1,2,3,4,5,6,7,8,9,10,11,12,13,14,15);
        v16h b1 = __builtin_shufflevector(*(const v8h*)(B1 + s * 32), *(const v8h*)(B1 + s * 32 + 16),
                    0,1,2,3,4,5,6,7,8,9,10,11,12,13,14,15);

        acc00 = __builtin_amdgcn_wmma_f32_16x16x32_f16(false, a0, false, b0,
                    (short)0, acc00, false, false);
        acc01 = __builtin_amdgcn_wmma_f32_16x16x32_f16(false, a0, false, b1,
                    (short)0, acc01, false, false);
        acc10 = __builtin_amdgcn_wmma_f32_16x16x32_f16(false, a1, false, b0,
                    (short)0, acc10, false, false);
        acc11 = __builtin_amdgcn_wmma_f32_16x16x32_f16(false, a1, false, b1,
                    (short)0, acc11, false, false);
    }
    __syncthreads();                              // tile region reused as C scratch

    // ---- epilogue: transpose each 16x16 f32 tile through LDS, then
    //      64B-contiguous global stores with bias add ----
    float* Cw = ((float*)Ah) + wv * (16 * 17);    // 272 floats per wave
    auto epi = [&](v8f acc, int mt, int ct) {
#pragma unroll
        for (int r = 0; r < 8; ++r) {
            // acc[r] sits at M = r + 8*g (pixel), N = mrow (co offset)
            Cw[(r + 8 * g) * 17 + mrow] = acc[r];
        }
        float* outp = out + (((size_t)n * COUT + co0 + ct * 16) * IH + y) * IW
                          + x0 + mt * 16;
#pragma unroll
        for (int r2 = 0; r2 < 8; ++r2) {
            int cc  = 2 * r2 + g;                 // co offset for this half-wave
            float v = Cw[mrow * 17 + cc] + bias[co0 + ct * 16 + cc];
            outp[(size_t)cc * IH * IW + mrow] = v;
        }
    };
    epi(acc00, 0, 0);
    epi(acc01, 0, 1);
    epi(acc10, 1, 0);
    epi(acc11, 1, 1);
}

// ---------------------------------------------------------------------------
extern "C" void kernel_launch(void* const* d_in, const int* in_sizes, int n_in,
                              void* d_out, int out_size, void* d_ws, size_t ws_size,
                              hipStream_t stream) {
    const float* x    = (const float*)d_in[0];
    const float* w    = (const float*)d_in[1];
    const float* bias = (const float*)d_in[2];
    float* out        = (float*)d_out;

    float*    ws_delta = (float*)d_ws;
    _Float16* wt       = (_Float16*)((char*)d_ws + 64);   // 589824 B ternary f16 weights

    delta_kernel<<<1, 256, 0, stream>>>(w, ws_delta);
    tern_kernel<<<WELEMS / 256, 256, 0, stream>>>(w, ws_delta, wt);

    // 32 n * 64 y * 2 x-halves = 4096 blocks, each 32 px x 256 co
    conv_wmma_kernel<<<4096, 256, 0, stream>>>(x, wt, bias, out);
}
